// DynamicFNO1d_51376398795162
// MI455X (gfx1250) — compile-verified, matrix operand order checked by reading, not run on hardware
//
#include <hip/hip_runtime.h>
#include <hip/hip_bf16.h>

typedef __attribute__((ext_vector_type(16))) _Float16 v16h;
typedef __attribute__((ext_vector_type(8)))  float    v8f;
typedef __attribute__((ext_vector_type(4))) unsigned int u32x4;
typedef __attribute__((ext_vector_type(8))) int          i32x8;
typedef __attribute__((ext_vector_type(4))) int          i32x4;

#define B_N 32
#define S_N 8192
#define W_N 64
#define MODES_N 16
#define L_N 20
#define M_N 8
#define PI_F 3.14159265358979f

__device__ __forceinline__ float gelu_exact(float x) {
  return 0.5f * x * (1.0f + erff(x * 0.7071067811865476f));
}

// ------------- trig basis table: tabT[n][k], n<16: cos(w0*n*k), n>=16: sin(w0*(n-16)*k) -------------
__global__ void basis_kernel(_Float16* __restrict__ tab) {
  int idx = blockIdx.x * blockDim.x + threadIdx.x;   // 32*8192 entries
  int n = idx >> 13;
  int k = idx & (S_N - 1);
  const float w0 = 2.f * PI_F / (float)S_N;
  float v;
  if (n < 16) v = __cosf(w0 * (float)n * (float)k);
  else        v = __sinf(w0 * (float)(n - 16) * (float)k);
  tab[idx] = (_Float16)v;
}

// ---------------- fc0: [x, grid] @ fc0_w + b -> h [B,W,S] (f16) ----------------
__global__ void fc0_kernel(const float* __restrict__ x,
                           const float* __restrict__ fc0_w,
                           const float* __restrict__ fc0_b,
                           _Float16* __restrict__ h) {
  int b = blockIdx.y;
  int s = blockIdx.x * blockDim.x + threadIdx.x;
  float xv = x[(size_t)b * S_N + s];
  float gv = (float)s * (1.0f / (float)(S_N - 1));
  for (int c = 0; c < W_N; ++c) {
    float v = xv * fc0_w[c] + gv * fc0_w[W_N + c] + fc0_b[c];
    h[((size_t)(b * W_N + c)) * S_N + s] = (_Float16)v;
  }
}

// ---------- per-layer mode-sums: S[i,o] = sum_y w[i,o,y] (complex) ----------
__global__ void prep_sums_kernel(const float* __restrict__ ar, const float* __restrict__ ai,
                                 const float* __restrict__ br, const float* __restrict__ bi,
                                 float* __restrict__ SAr, float* __restrict__ SAi,
                                 float* __restrict__ SBr, float* __restrict__ SBi) {
  int idx = blockIdx.x * blockDim.x + threadIdx.x;  // i*64 + o
  float s0 = 0.f, s1 = 0.f, s2 = 0.f, s3 = 0.f;
  for (int y = 0; y < MODES_N; ++y) {
    s0 += ar[idx * MODES_N + y];
    s1 += ai[idx * MODES_N + y];
    s2 += br[idx * MODES_N + y];
    s3 += bi[idx * MODES_N + y];
  }
  SAr[idx] = s0; SAi[idx] = s1; SBr[idx] = s2; SBi[idx] = s3;
}

// ---------------- CFT gate: spec(16 bins) -> |.| mean -> sigmoid ----------------
__global__ void gate_kernel(const _Float16* __restrict__ h,
                            const float* __restrict__ gate_w,  // [16,64]
                            const float* __restrict__ gate_b,  // [64]
                            float* __restrict__ gate) {        // [B,64]
  __shared__ float cwr[M_N][MODES_N];
  __shared__ float cwi[M_N][MODES_N];
  __shared__ float magArr[W_N][MODES_N];
  __shared__ float magf[MODES_N];
  int b = blockIdx.x;
  int c = threadIdx.x;  // 64 threads

  for (int e = c; e < M_N * MODES_N; e += W_N) {
    int m = e / MODES_N, f = e % MODES_N;
    float sr = 0.f, si = 0.f;
    for (int j = 0; j < M_N; ++j) {
      float half = (2.f * j + 1.f) * PI_F / (2.f * M_N);
      float cj = -__cosf(half);
      float ac = PI_F - half;                 // arccos(cj)
      float T = __cosf((float)m * ac);
      float ang = cj * (float)f * (PI_F / (float)L_N);
      float sv, cv; __sincosf(ang, &sv, &cv);
      sr += T * cv;
      si -= T * sv;
    }
    cwr[m][f] = sr * (1.f / (2.f * (float)L_N));
    cwi[m][f] = si * (1.f / (2.f * (float)L_N));
  }
  __syncthreads();

  float spr[MODES_N], spi[MODES_N];
  for (int f = 0; f < MODES_N; ++f) { spr[f] = 0.f; spi[f] = 0.f; }
  const _Float16* row = h + (size_t)(b * W_N + c) * S_N;
  for (int l = 0; l < L_N; ++l) {
    float start = (float)l / (float)L_N;
    float segr[MODES_N], segi[MODES_N];
    for (int f = 0; f < MODES_N; ++f) { segr[f] = 0.f; segi[f] = 0.f; }
    for (int m = 0; m < M_N; ++m) {
      float cm = -__cosf((2.f * m + 1.f) * PI_F / (2.f * M_N));
      float tseg = start + (cm + 1.f) * (0.5f / (float)L_N);
      float pos = tseg * (float)(S_N - 1);
      int ridx = (int)ceilf(pos);
      ridx = min(max(ridx, 0), S_N - 1);
      int lidx = max(ridx - 1, 0);
      float tl = (float)lidx * (1.f / (float)(S_N - 1));
      float tr = (float)ridx * (1.f / (float)(S_N - 1));
      float den = (tr - tl == 0.f) ? 1.f : (tr - tl);
      float wr = (tseg - tl) / den;
      float sseg = (1.f - wr) * (float)row[lidx] + wr * (float)row[ridx];
      for (int f = 0; f < MODES_N; ++f) {
        segr[f] += sseg * cwr[m][f];
        segi[f] += sseg * cwi[m][f];
      }
    }
    for (int f = 0; f < MODES_N; ++f) {
      float shift = 2.f * PI_F * (float)f * start;
      float esi, esr; __sincosf(shift, &esi, &esr);
      esi = -esi;
      spr[f] += esr * segr[f] - esi * segi[f];
      spi[f] += esr * segi[f] + esi * segr[f];
    }
  }
  for (int f = 0; f < MODES_N; ++f)
    magArr[c][f] = sqrtf(spr[f] * spr[f] + spi[f] * spi[f]);
  __syncthreads();
  if (c < MODES_N) {
    float s = 0.f;
    for (int cc = 0; cc < W_N; ++cc) s += magArr[cc][c];
    magf[c] = s * (1.f / (float)W_N);
  }
  __syncthreads();
  float acc = gate_b[c];
  for (int f = 0; f < MODES_N; ++f) acc += magf[f] * gate_w[f * W_N + c];
  gate[b * W_N + c] = 1.f / (1.f + __expf(-acc));
}

// --------- forward DFT (first 16 bins) as WMMA GEMM: K=8192, N=32 ---------
__global__ void hft_kernel(const _Float16* __restrict__ h,
                           const _Float16* __restrict__ tab,
                           float* __restrict__ hft) {
  int tid = threadIdx.x;
  int wv = tid >> 5;
  int lane = tid & 31;
  int r0 = (blockIdx.x * 8 + wv) * 16;
  int row = r0 + (lane & 15);
  int khalf = (lane >> 4) * 16;
  const _Float16* rp = h + (size_t)row * S_N;
  const _Float16* tc = tab + (size_t)(lane & 15) * S_N;
  const _Float16* ts = tab + (size_t)(16 + (lane & 15)) * S_N;
  v8f acc_re = {};
  v8f acc_im = {};
  for (int k0 = 0; k0 < S_N; k0 += 32) {
    __builtin_prefetch(rp + k0 + khalf + 1024, 0, 3);
    v16h a  = *reinterpret_cast<const v16h*>(rp + k0 + khalf);
    v16h bc = *reinterpret_cast<const v16h*>(tc + k0 + khalf);
    v16h bs = *reinterpret_cast<const v16h*>(ts + k0 + khalf);
    acc_re = __builtin_amdgcn_wmma_f32_16x16x32_f16(false, a, false, bc, (short)0, acc_re, false, false);
    acc_im = __builtin_amdgcn_wmma_f32_16x16x32_f16(false, a, false, bs, (short)0, acc_im, false, false);
  }
  int n = lane & 15;
  int mh = (lane >> 4) * 8;
  for (int r = 0; r < 8; ++r) {
    int gr = r0 + r + mh;
    hft[gr * 32 + n]      = acc_re[r];
    hft[gr * 32 + 16 + n] = -acc_im[r];
  }
}

// ----- per-batch mode mix + build fused f16 A-matrix Acomb[b][o][96] -----
__global__ void modes_kernel(const float* __restrict__ hft,
                             const float* __restrict__ gate,
                             const float* __restrict__ SAr, const float* __restrict__ SAi,
                             const float* __restrict__ SBr, const float* __restrict__ SBi,
                             const float* __restrict__ ww,
                             _Float16* __restrict__ Acomb) {
  int b = blockIdx.x;
  int o = threadIdx.x;
  float g = gate[b * W_N + o];
  _Float16* arow = Acomb + (size_t)b * W_N * 96 + (size_t)o * 96;
  for (int c = 0; c < W_N; ++c) arow[c] = (_Float16)ww[o * W_N + c];
  for (int f = 0; f < MODES_N; ++f) {
    float par = 0.f, pai = 0.f, pbr = 0.f, pbi = 0.f;
    for (int i = 0; i < W_N; ++i) {
      float hr = hft[(b * W_N + i) * 32 + f];
      float hi = hft[(b * W_N + i) * 32 + 16 + f];
      float sar = SAr[i * W_N + o], sai = SAi[i * W_N + o];
      float sbr = SBr[i * W_N + o], sbi = SBi[i * W_N + o];
      par += hr * sar - hi * sai;  pai += hr * sai + hi * sar;
      pbr += hr * sbr - hi * sbi;  pbi += hr * sbi + hi * sbr;
    }
    float omr = g * par + (1.f - g) * pbr;
    float omi = g * pai + (1.f - g) * pbi;
    float sc = (f == 0) ? (1.f / (float)S_N) : (2.f / (float)S_N);
    arow[64 + f] = (_Float16)(sc * omr);
    arow[80 + f] = (_Float16)((f == 0) ? 0.f : (-2.f / (float)S_N) * omi);
  }
}

// --------- fused irfft(16 modes) + 1x1 conv + bias + GELU, WMMA K=96 ---------
// A-matrix (64x96 f16, 12KB, contiguous) is DMA'd into LDS via the Tensor Data Mover.
__global__ void layer_kernel(const _Float16* __restrict__ h,
                             const _Float16* __restrict__ tab,
                             const _Float16* __restrict__ Acomb,
                             const float* __restrict__ wb,
                             _Float16* __restrict__ hout,
                             int apply_gelu) {
  __shared__ __align__(32) _Float16 As[W_N][96];    // A: [o][k] -- filled by TDM
  __shared__ __align__(32) _Float16 BsT[64][96];    // B^T: [s-col][k]
  int b = blockIdx.y;
  int s0 = blockIdx.x * 64;
  int tid = threadIdx.x;
  int wv = tid >> 5, lane = tid & 31;

  if (wv == 0) {
    // Tensor DMA descriptor: 1 tile, 2D tensor (6144 x 1 elements of 2 bytes)
    unsigned long long ga =
        (unsigned long long)(const void*)(Acomb + (size_t)b * W_N * 96);
    unsigned lds_off = (unsigned)(unsigned long long)
        (__attribute__((address_space(3))) _Float16*)&As[0][0];
    u32x4 g0;
    g0[0] = 1u;                                   // count=1, user mode
    g0[1] = lds_off;                              // lds_addr (bytes)
    g0[2] = (unsigned)(ga & 0xFFFFFFFFull);       // global_addr[31:0]
    g0[3] = (unsigned)((ga >> 32) & 0x01FFFFFFull) | (2u << 30); // addr[56:32] | type=2
    i32x8 g1;
    g1[0] = 0x00010000;                           // workgroup_mask=0, data_size=1 (2B)
    g1[1] = (int)((6144u & 0xFFFFu) << 16);       // tensor_dim0[15:0]
    g1[2] = (int)((6144u >> 16) | (1u << 16));    // tensor_dim0[31:16] | tensor_dim1[15:0]=1
    g1[3] = (int)(6144u << 16);                   // tile_dim0 = 6144
    g1[4] = 1;                                    // tile_dim1 = 1, tile_dim2 = 0
    g1[5] = 6144;                                 // tensor_dim0_stride[31:0]
    g1[6] = 0;
    g1[7] = 0;
    i32x4 gz4 = {0, 0, 0, 0};
    i32x8 gz8 = {0, 0, 0, 0, 0, 0, 0, 0};
    __builtin_amdgcn_tensor_load_to_lds(g0, g1, gz4, gz4, gz8, 0);
    __builtin_amdgcn_s_wait_tensorcnt(0);
  }

  int n = tid & 63;
  int q = tid >> 6;
  int s = s0 + n;
  for (int k = q; k < 64; k += 4)
    BsT[n][k] = h[(size_t)(b * W_N + k) * S_N + s];
  for (int k = 64 + q; k < 96; k += 4)
    BsT[n][k] = tab[(size_t)(k - 64) * S_N + s];
  __syncthreads();

  int lr = lane & 15, lh = lane >> 4;
  for (int tt = 0; tt < 2; ++tt) {
    int tile = wv * 2 + tt;         // 16 tiles: 4(M) x 4(N)
    int mi = tile >> 2, ni = tile & 3;
    v8f acc = {};
    for (int ch = 0; ch < 3; ++ch) {
      int ko = ch * 32 + lh * 16;
      v16h a  = *reinterpret_cast<const v16h*>(&As[mi * 16 + lr][ko]);
      v16h bb = *reinterpret_cast<const v16h*>(&BsT[ni * 16 + lr][ko]);
      acc = __builtin_amdgcn_wmma_f32_16x16x32_f16(false, a, false, bb, (short)0, acc, false, false);
    }
    for (int r = 0; r < 8; ++r) {
      int o = mi * 16 + r + lh * 8;
      int sc = s0 + ni * 16 + lr;
      float v = acc[r] + wb[o];
      if (apply_gelu) v = gelu_exact(v);
      hout[(size_t)(b * W_N + o) * S_N + sc] = (_Float16)v;
    }
  }
}

// --------------- head: gelu(h^T @ fc1 + b) @ fc2 + b, WMMA ---------------
__global__ void head_kernel(const _Float16* __restrict__ h,
                            const float* __restrict__ fc1_w,  // [64,128]
                            const float* __restrict__ fc1_b,  // [128]
                            const float* __restrict__ fc2_w,  // [128]
                            const float* __restrict__ fc2_b,  // [1]
                            float* __restrict__ out) {        // [B,S]
  __shared__ __align__(32) _Float16 As[64][64];
  __shared__ __align__(32) _Float16 BT[128][64];
  __shared__ float zs[64][128];
  int b = blockIdx.y;
  int s0 = blockIdx.x * 64;
  int tid = threadIdx.x;
  int n = tid & 63, q = tid >> 6;
  for (int c = q; c < 64; c += 4)
    As[n][c] = h[(size_t)(b * W_N + c) * S_N + s0 + n];
  for (int idx = tid; idx < 128 * 64; idx += 256) {
    int j = idx >> 6, c = idx & 63;
    BT[j][c] = (_Float16)fc1_w[c * 128 + j];
  }
  __syncthreads();
  int wv = tid >> 5, lane = tid & 31;
  int lr = lane & 15, lh = lane >> 4;
  for (int tt = 0; tt < 4; ++tt) {
    int tile = wv * 4 + tt;        // 32 tiles: 4(M) x 8(N)
    int mi = tile & 3, ni = tile >> 2;
    v8f acc = {};
    for (int ch = 0; ch < 2; ++ch) {
      int ko = ch * 32 + lh * 16;
      v16h a  = *reinterpret_cast<const v16h*>(&As[mi * 16 + lr][ko]);
      v16h bb = *reinterpret_cast<const v16h*>(&BT[ni * 16 + lr][ko]);
      acc = __builtin_amdgcn_wmma_f32_16x16x32_f16(false, a, false, bb, (short)0, acc, false, false);
    }
    for (int r = 0; r < 8; ++r) {
      int sl = mi * 16 + r + lh * 8;
      int j = ni * 16 + lr;
      zs[sl][j] = gelu_exact(acc[r] + fc1_b[j]);
    }
  }
  __syncthreads();
  if (tid < 64) {
    float a = fc2_b[0];
    for (int j = 0; j < 128; ++j) a += zs[tid][j] * fc2_w[j];
    out[(size_t)b * S_N + s0 + tid] = a;
  }
}

extern "C" void kernel_launch(void* const* d_in, const int* in_sizes, int n_in,
                              void* d_out, int out_size, void* d_ws, size_t ws_size,
                              hipStream_t stream) {
  const float* x      = (const float*)d_in[0];
  const float* fc0_w  = (const float*)d_in[1];
  const float* fc0_b  = (const float*)d_in[2];
  const float* gate_w = (const float*)d_in[3];
  const float* gate_b = (const float*)d_in[4];
  const float* cAre   = (const float*)d_in[5];
  const float* cAim   = (const float*)d_in[6];
  const float* cBre   = (const float*)d_in[7];
  const float* cBim   = (const float*)d_in[8];
  const float* ww     = (const float*)d_in[9];
  const float* wb     = (const float*)d_in[10];
  const float* fc1_w  = (const float*)d_in[11];
  const float* fc1_b  = (const float*)d_in[12];
  const float* fc2_w  = (const float*)d_in[13];
  const float* fc2_b  = (const float*)d_in[14];
  float* out = (float*)d_out;

  char* ws = (char*)d_ws;
  const size_t HSZ = (size_t)B_N * W_N * S_N * sizeof(_Float16);  // 32 MB
  _Float16* hA  = (_Float16*)(ws);
  _Float16* hB  = (_Float16*)(ws + HSZ);
  _Float16* tab = (_Float16*)(ws + 2 * HSZ);                       // 512 KB
  char* p = ws + 2 * HSZ + (size_t)32 * S_N * sizeof(_Float16);
  float* hft   = (float*)p;                                        // [2048][32]
  float* gateB = hft + (size_t)B_N * W_N * 32;
  float* SAr   = gateB + B_N * W_N;
  float* SAi   = SAr + W_N * W_N;
  float* SBr   = SAi + W_N * W_N;
  float* SBi   = SBr + W_N * W_N;
  _Float16* Acomb = (_Float16*)(SBi + W_N * W_N);  // [B][64][96] f16

  basis_kernel<<<(32 * S_N) / 256, 256, 0, stream>>>(tab);
  fc0_kernel<<<dim3(S_N / 256, B_N), 256, 0, stream>>>(x, fc0_w, fc0_b, hA);

  _Float16* cur = hA;
  _Float16* nxt = hB;
  for (int l = 0; l < 4; ++l) {
    int wsz = W_N * W_N * MODES_N;
    prep_sums_kernel<<<16, 256, 0, stream>>>(cAre + l * wsz, cAim + l * wsz,
                                             cBre + l * wsz, cBim + l * wsz,
                                             SAr, SAi, SBr, SBi);
    gate_kernel<<<B_N, W_N, 0, stream>>>(cur, gate_w + l * MODES_N * W_N,
                                         gate_b + l * W_N, gateB);
    hft_kernel<<<16, 256, 0, stream>>>(cur, tab, hft);
    modes_kernel<<<B_N, W_N, 0, stream>>>(hft, gateB, SAr, SAi, SBr, SBi,
                                          ww + l * W_N * W_N, Acomb);
    layer_kernel<<<dim3(S_N / 64, B_N), 256, 0, stream>>>(cur, tab, Acomb,
                                                          wb + l * W_N, nxt,
                                                          (l < 3) ? 1 : 0);
    _Float16* t = cur; cur = nxt; nxt = t;
  }
  head_kernel<<<dim3(S_N / 64, B_N), 256, 0, stream>>>(cur, fc1_w, fc1_b, fc2_w, fc2_b, out);
}
